// MultiHeadAttention_57561151701296
// MI455X (gfx1250) — matrix operand order repaired and checked by reading, not compiled, not run on hardware
//
#include <hip/hip_runtime.h>

#define DIM    768
#define NHEADS 16
#define HDIM   48
#define HPAD   64
#define BSZ    2
#define SEQ    2048
#define MTOT   (BSZ * SEQ)   // 4096

static __device__ __constant__ float kScale = 0.14433756729740643f; // 48^-0.5

typedef __attribute__((ext_vector_type(16))) _Float16 v16h;
typedef __attribute__((ext_vector_type(8)))  _Float16 v8h;
typedef __attribute__((ext_vector_type(8)))  float    v8f;

// ---- WMMA fragment helpers (wave32, 16x16x32 f16 -> f32) -------------------
// A-frag (MxK=16x32): lane(lo,hi) holds row M=lo, K = {0..7,16..23} + 8*hi
__device__ __forceinline__ v16h load_a_frag(const _Float16* row, int k0, int hi) {
  v8h a0 = *(const v8h*)(row + k0 + 8 * hi);
  v8h a1 = *(const v8h*)(row + k0 + 16 + 8 * hi);
  v16h r;
#pragma unroll
  for (int i = 0; i < 8; ++i) { r[i] = a0[i]; r[i + 8] = a1[i]; }
  return r;
}
// B-frag (KxN=32x16): lane(lo,hi) holds col N=lo, K = {0..7,16..23} + 8*hi (same as A)
__device__ __forceinline__ v16h load_b_frag(const _Float16* row, int k0, int hi) {
  return load_a_frag(row, k0, hi);
}
__device__ __forceinline__ v8f wmma(v16h a, v16h b, v8f c) {
  return __builtin_amdgcn_wmma_f32_16x16x32_f16(false, a, false, b, (short)0, c,
                                                false, false);
}

// ---- elementwise helpers ---------------------------------------------------
__global__ void cvt_f32_to_f16(const float* __restrict__ src,
                               _Float16* __restrict__ dst, int n) {
  int i = blockIdx.x * blockDim.x + threadIdx.x;
  if (i < n) dst[i] = (_Float16)src[i];
}

__global__ void zfill_f16(_Float16* __restrict__ p, int n) {
  int i = blockIdx.x * blockDim.x + threadIdx.x;
  if (i < n) p[i] = (_Float16)0.f;
}

// ---- GEMM: C[m][n] = sum_k A[m][k] * Bt[n][k]  (M=4096, N=768, K=768) ------
// mode 0: fp32 to outF[m*DIM+n]
// mode 1: f16 to per-head padded buffer outQ[b][h][s][HPAD]  (n = h*48+d)
// mode 2: f16 TRANSPOSED to outQ[b][h][d][s]  (d-major V for the PV B-frags);
//         D-frag regs are 8 consecutive s at fixed d -> single v8h store
__global__ __launch_bounds__(256) void gemm_f16(const _Float16* __restrict__ A,
                                                const _Float16* __restrict__ Bt,
                                                float* __restrict__ outF,
                                                _Float16* __restrict__ outQ,
                                                int mode) {
  const int lane = threadIdx.x & 31;
  const int wave = threadIdx.x >> 5;
  const int lo = lane & 15, hi = lane >> 4;
  const int mbase = blockIdx.x * 64 + (wave >> 1) * 16;
  const int nbase = blockIdx.y * 128 + (wave & 1) * 64;

  const _Float16* arow = A + (size_t)(mbase + lo) * DIM;
  v8f acc[4] = {};
  for (int k0 = 0; k0 < DIM; k0 += 32) {
    v16h af = load_a_frag(arow, k0, hi);
#pragma unroll
    for (int t = 0; t < 4; ++t) {
      v16h bf = load_b_frag(Bt + (size_t)(nbase + 16 * t + lo) * DIM, k0, hi);
      acc[t] = wmma(af, bf, acc[t]);
    }
  }
#pragma unroll
  for (int t = 0; t < 4; ++t) {
    const int n = nbase + 16 * t + lo;
    if (mode == 2) {
      const int h = n / HDIM, d = n % HDIM;
      const int b = mbase / SEQ;              // tile never crosses batch
      const int s0 = mbase - b * SEQ + 8 * hi;
      v8h pk;
#pragma unroll
      for (int r = 0; r < 8; ++r) pk[r] = (_Float16)acc[t][r];
      *(v8h*)(outQ + ((size_t)((b * NHEADS + h) * HDIM + d)) * SEQ + s0) = pk;
    } else {
#pragma unroll
      for (int r = 0; r < 8; ++r) {
        const int m = mbase + r + 8 * hi;
        if (mode == 0) {
          outF[(size_t)m * DIM + n] = acc[t][r];
        } else {
          const int h = n / HDIM, d = n % HDIM;
          const int b = m / SEQ, s = m % SEQ;
          outQ[(((size_t)(b * NHEADS + h) * SEQ + s) * HPAD) + d] =
              (_Float16)acc[t][r];
        }
      }
    }
  }
}

// ---- Flash attention: one wave = 16 queries of one (b,h) -------------------
// Q,K: [B*H][SEQ][HPAD] f16 (d pads zeroed).  Vt: [B*H][HDIM][SEQ] f16.
__global__ __launch_bounds__(256) void attn_kernel(const _Float16* __restrict__ Q,
                                                   const _Float16* __restrict__ K,
                                                   const _Float16* __restrict__ Vt,
                                                   _Float16* __restrict__ ctx) {
  const int lane = threadIdx.x & 31;
  const int wave = threadIdx.x >> 5;
  const int lo = lane & 15, hi = lane >> 4;
  const int bh = blockIdx.x;                       // 0..B*H-1
  const int qbase = blockIdx.y * 128 + wave * 16;  // 16-query tile

  const _Float16* Qh = Q + (size_t)bh * SEQ * HPAD;
  const _Float16* Kh = K + (size_t)bh * SEQ * HPAD;
  const _Float16* Vh = Vt + (size_t)bh * HDIM * SEQ;
  // per-lane V^T rows for the 3 d-chunks (d = dc*16 + lo)
  const _Float16* vrow0 = Vh + (size_t)(0 * 16 + lo) * SEQ;
  const _Float16* vrow1 = Vh + (size_t)(1 * 16 + lo) * SEQ;
  const _Float16* vrow2 = Vh + (size_t)(2 * 16 + lo) * SEQ;

  // Q as B-fragment (Qt): b[e] = Q[qbase+lo][d = {0..7,16..23} + 8*hi (+32)]
  const _Float16* qrow = Qh + (size_t)(qbase + lo) * HPAD;
  const v16h qb0 = load_a_frag(qrow, 0, hi);
  const v16h qb1 = load_a_frag(qrow, 32, hi);

  float m_run = -1e30f, l_run = 0.f;
  v8f c0 = {}, c1 = {}, c2 = {};

  for (int kv0 = 0; kv0 < SEQ; kv0 += 32) {
    // prefetch next kv step while this one computes
    if (kv0 + 32 < SEQ) {
      __builtin_prefetch(Kh + (size_t)(kv0 + 32 + lo) * HPAD, 0, 1);
      __builtin_prefetch(Kh + (size_t)(kv0 + 48 + lo) * HPAD, 0, 1);
      __builtin_prefetch(vrow0 + kv0 + 32, 0, 1);
      __builtin_prefetch(vrow1 + kv0 + 32, 0, 1);
      __builtin_prefetch(vrow2 + kv0 + 32, 0, 1);
    }

    // S^T tiles: D[kv][q] = sum_d K[kv][d] * Q[q][d]
    const _Float16* kr1 = Kh + (size_t)(kv0 + lo) * HPAD;
    const _Float16* kr2 = kr1 + 16 * HPAD;
    v8f s1 = {}, s2 = {};
    s1 = wmma(load_a_frag(kr1, 0, hi), qb0, s1);
    s1 = wmma(load_a_frag(kr1, 32, hi), qb1, s1);
    s2 = wmma(load_a_frag(kr2, 0, hi), qb0, s2);
    s2 = wmma(load_a_frag(kr2, 32, hi), qb1, s2);

    // online softmax for query q=lo (16 kv here + 16 in partner lane lo^16)
    float mt = -1e30f;
#pragma unroll
    for (int r = 0; r < 8; ++r) {
      s1[r] *= kScale; s2[r] *= kScale;
      mt = fmaxf(mt, fmaxf(s1[r], s2[r]));
    }
    mt = fmaxf(mt, __shfl_xor(mt, 16, 32));
    const float m_new = fmaxf(m_run, mt);

    float sum = 0.f;
    v16h pa;  // A-frag of P: a[e] = P[q=lo][kv] -- exactly our score regs
#pragma unroll
    for (int r = 0; r < 8; ++r) {
      const float p1 = __expf(s1[r] - m_new);
      const float p2 = __expf(s2[r] - m_new);
      sum += p1 + p2;
      pa[r] = (_Float16)p1;
      pa[r + 8] = (_Float16)p2;
    }
    sum += __shfl_xor(sum, 16, 32);
    const float corr = __expf(m_run - m_new);
    l_run = l_run * corr + sum;
    m_run = m_new;

    // rescale running context: ctx reg r belongs to query (r + 8*hi)
#pragma unroll
    for (int r = 0; r < 8; ++r) {
      const float cr = __shfl(corr, r + 8 * hi, 32);
      c0[r] *= cr; c1[r] *= cr; c2[r] *= cr;
    }

    // P(16x32) x V chunks; V B-frag is two 16B loads from V^T per chunk:
    // b[e] = V^T[dc*16+lo][kv0 + {0..7,16..23} + 8*hi]
    c0 = wmma(pa, load_a_frag(vrow0, kv0, hi), c0);
    c1 = wmma(pa, load_a_frag(vrow1, kv0, hi), c1);
    c2 = wmma(pa, load_a_frag(vrow2, kv0, hi), c2);
  }

  // epilogue: ctx[q][d] / l_q  ->  packed f16 [B,S,DIM] for the Wo GEMM
  const int b = bh >> 4, h = bh & 15;
#pragma unroll
  for (int r = 0; r < 8; ++r) {
    const float lr = __shfl(l_run, r + 8 * hi, 32);
    const float inv = 1.f / lr;
    const int s = qbase + r + 8 * hi;
    _Float16* dst = ctx + ((size_t)(b * SEQ + s)) * DIM + h * HDIM;
    dst[0 * 16 + lo] = (_Float16)(c0[r] * inv);
    dst[1 * 16 + lo] = (_Float16)(c1[r] * inv);
    dst[2 * 16 + lo] = (_Float16)(c2[r] * inv);
  }
}

// ---------------------------------------------------------------------------
extern "C" void kernel_launch(void* const* d_in, const int* in_sizes, int n_in,
                              void* d_out, int out_size, void* d_ws,
                              size_t ws_size, hipStream_t stream) {
  (void)in_sizes; (void)n_in; (void)out_size; (void)ws_size;
  const float* x  = (const float*)d_in[0];
  const float* Wq = (const float*)d_in[1];
  const float* Wk = (const float*)d_in[2];
  const float* Wv = (const float*)d_in[3];
  const float* Wo = (const float*)d_in[4];
  float* out = (float*)d_out;

  char* ws = (char*)d_ws;
  size_t off = 0;
  auto alloc = [&](size_t bytes) -> void* {
    void* p = ws + off;
    off += (bytes + 255) & ~(size_t)255;
    return p;
  };
  const size_t nX = (size_t)MTOT * DIM;                 // 3.1M elems
  const size_t nW = (size_t)DIM * DIM;                  // 0.59M elems
  const size_t nH = (size_t)BSZ * NHEADS * SEQ * HPAD;  // 4.2M elems (padded)
  const size_t nV = (size_t)BSZ * NHEADS * HDIM * SEQ;  // 3.1M elems (V^T)
  _Float16* xh   = (_Float16*)alloc(nX * 2);
  _Float16* wqh  = (_Float16*)alloc(nW * 2);
  _Float16* wkh  = (_Float16*)alloc(nW * 2);
  _Float16* wvh  = (_Float16*)alloc(nW * 2);
  _Float16* woh  = (_Float16*)alloc(nW * 2);
  _Float16* qb   = (_Float16*)alloc(nH * 2);
  _Float16* kb   = (_Float16*)alloc(nH * 2);
  _Float16* vt   = (_Float16*)alloc(nV * 2);
  _Float16* ctxh = (_Float16*)alloc(nX * 2);

  const int T = 256;
  cvt_f32_to_f16<<<(int)((nX + T - 1) / T), T, 0, stream>>>(x, xh, (int)nX);
  cvt_f32_to_f16<<<(int)((nW + T - 1) / T), T, 0, stream>>>(Wq, wqh, (int)nW);
  cvt_f32_to_f16<<<(int)((nW + T - 1) / T), T, 0, stream>>>(Wk, wkh, (int)nW);
  cvt_f32_to_f16<<<(int)((nW + T - 1) / T), T, 0, stream>>>(Wv, wvh, (int)nW);
  cvt_f32_to_f16<<<(int)((nW + T - 1) / T), T, 0, stream>>>(Wo, woh, (int)nW);

  // zero head-dim pads of Q and K (so the K=64 WMMA reduction is exact)
  zfill_f16<<<(int)((nH + T - 1) / T), T, 0, stream>>>(qb, (int)nH);
  zfill_f16<<<(int)((nH + T - 1) / T), T, 0, stream>>>(kb, (int)nH);

  dim3 gemmGrid(MTOT / 64, DIM / 128);  // 64 x 6
  gemm_f16<<<gemmGrid, 256, 0, stream>>>(xh, wqh, nullptr, qb, 1);
  gemm_f16<<<gemmGrid, 256, 0, stream>>>(xh, wkh, nullptr, kb, 1);
  gemm_f16<<<gemmGrid, 256, 0, stream>>>(xh, wvh, nullptr, vt, 2);

  attn_kernel<<<dim3(BSZ * NHEADS, SEQ / 128), 256, 0, stream>>>(qb, kb, vt,
                                                                 ctxh);

  gemm_f16<<<gemmGrid, 256, 0, stream>>>(ctxh, woh, out, nullptr, 0);
}
